// MetricSelfAttention_21784074125668
// MI455X (gfx1250) — compile-verified, matrix operand order checked by reading, not run on hardware
//
#include <hip/hip_runtime.h>
#include <hip/hip_bf16.h>

typedef __bf16 v16bf __attribute__((ext_vector_type(16)));
typedef __bf16 v8bf  __attribute__((ext_vector_type(8)));
typedef float  v8f   __attribute__((ext_vector_type(8)));
typedef int    v4i   __attribute__((ext_vector_type(4)));

#define WMMA_BF16(a,b,c) __builtin_amdgcn_wmma_f32_16x16x32_bf16(false,(a),false,(b),(short)0,(c),false,false)

#if defined(__has_builtin)
#if __has_builtin(__builtin_amdgcn_global_load_async_to_lds_b128) && \
    __has_builtin(__builtin_amdgcn_s_wait_asynccnt)
#define HAVE_ASYNC_LDS 1
#endif
#endif

__device__ __forceinline__ v16bf make_frag(const __bf16* p0, const __bf16* p1) {
  v8bf lo = *(const v8bf*)p0;
  v8bf hi = *(const v8bf*)p1;
  v16bf r;
#pragma unroll
  for (int i = 0; i < 8; i++) { r[i] = lo[i]; r[8 + i] = hi[i]; }
  return r;
}

// ---------------- fp32 -> bf16 convert ----------------
__global__ void k_cvt_bf16(const float* __restrict__ in, __bf16* __restrict__ out, int n) {
  int i = blockIdx.x * blockDim.x + threadIdx.x;
  if (i < n) out[i] = (__bf16)in[i];
}

// ---- shuffle W (row-major [N=1024, K=1024], B = W^T) into B-fragment-major bf16 ----
// layout: Bs[((tn*32 + ks)*32 + lane)*16 + e]
__global__ void k_shuffle_w(const float* __restrict__ W, __bf16* __restrict__ Bs) {
  int tid = blockIdx.x * blockDim.x + threadIdx.x;   // 1M threads
  int e    = tid & 15;
  int lane = (tid >> 4) & 31;
  int ks   = (tid >> 9) & 31;
  int tn   = tid >> 14;
  int half = lane >> 4;
  int ncol = tn * 16 + (lane & 15);
  int koff = (e < 8) ? (half * 8 + e) : (16 + half * 8 + (e - 8));
  int kk   = ks * 32 + koff;
  Bs[tid] = (__bf16)W[ncol * 1024 + kk];
}

// ---------------- per-head metric: softmax(tril(pm)/8) @ its transpose ----------------
__global__ void k_metric(const float* __restrict__ pm, __bf16* __restrict__ met) {
  __shared__ float P[64 * 64];
  int h = blockIdx.x, t = threadIdx.x;  // 64 threads
  const float* row = pm + h * 4096 + t * 64;
  float m = -3.0e38f;
  for (int j = 0; j <= t; j++) m = fmaxf(m, row[j]);
  float s = 0.f;
  for (int j = 0; j < 64; j++) {
    float p = (j <= t) ? __expf((row[j] - m) * 0.125f) : 0.f;
    P[t * 64 + j] = p; s += p;
  }
  float inv = 1.f / s;
  for (int j = 0; j < 64; j++) P[t * 64 + j] *= inv;
  __syncthreads();
  for (int j = 0; j < 64; j++) {
    float acc = 0.f;
    for (int k2 = 0; k2 < 64; k2++) acc += P[t * 64 + k2] * P[j * 64 + k2];
    met[h * 4096 + t * 64 + j] = (__bf16)acc;  // symmetric, layout-safe
  }
}

// ------- M=4096,N=1024,K=1024 bf16 WMMA GEMM; each wave: 16x64 strip (4 N-tiles) -------
__global__ void k_gemm(const __bf16* __restrict__ A, const __bf16* __restrict__ Bs,
                       float* __restrict__ outF, __bf16* __restrict__ outBF) {
  int lane = threadIdx.x & 31, wid = threadIdx.x >> 5;
  int half = lane >> 4, l15 = lane & 15;
  int tn0 = blockIdx.x * 4;        // 16 blocks in x -> 64 N-tiles
  int tm  = blockIdx.y * 8 + wid;  // 32 blocks in y -> 256 M-tiles
  const __bf16* Arow = A + (size_t)(tm * 16 + l15) * 1024;
  v8f c[4] = {v8f{}, v8f{}, v8f{}, v8f{}};
  for (int ks = 0; ks < 32; ks++) {
    int kb = ks * 32;
    __builtin_prefetch(Arow + kb + 256, 0, 1);   // global_prefetch_b8, ~8 ksteps ahead
    v16bf a = make_frag(Arow + kb + half * 8, Arow + kb + 16 + half * 8);
#pragma unroll
    for (int t = 0; t < 4; t++) {
      const __bf16* bp = Bs + (size_t)(((tn0 + t) * 32 + ks) * 32 + lane) * 16;
      v16bf b = make_frag(bp, bp + 8);
      c[t] = WMMA_BF16(a, b, c[t]);
    }
  }
#pragma unroll
  for (int t = 0; t < 4; t++) {
    int col = (tn0 + t) * 16 + l15;
#pragma unroll
    for (int r = 0; r < 8; r++) {
      int rowi = tm * 16 + r + 8 * half;
      if (outF)  outF[(size_t)rowi * 1024 + col] = c[t][r];
      if (outBF) outBF[(size_t)rowi * 1024 + col] = (__bf16)c[t][r];
    }
  }
}

// ---------------- qm[b,h] = proj_head[2048,64] @ metric[h][64,64] ----------------
__global__ void k_qm(const __bf16* __restrict__ projbf, const __bf16* __restrict__ met,
                     __bf16* __restrict__ qm) {
  int lane = threadIdx.x & 31, wid = threadIdx.x >> 5;
  int half = lane >> 4, l15 = lane & 15;
  int blk = blockIdx.x;            // 512
  int strip = blk & 15;
  int h = (blk >> 4) & 15;
  int b = blk >> 8;
  int row0 = strip * 128 + wid * 16;
  const __bf16* Arow = projbf + (size_t)(b * 2048 + row0 + l15) * 1024 + h * 64;
  v16bf a0 = make_frag(Arow + half * 8,      Arow + 16 + half * 8);
  v16bf a1 = make_frag(Arow + 32 + half * 8, Arow + 48 + half * 8);
  for (int tn = 0; tn < 4; tn++) {
    v8f c = {};
#pragma unroll
    for (int ks = 0; ks < 2; ks++) {
      v16bf bfg;
#pragma unroll
      for (int e = 0; e < 16; e++) {
        int koff = (e < 8) ? (half * 8 + e) : (16 + half * 8 + (e - 8));
        int kk = ks * 32 + koff;
        bfg[e] = met[h * 4096 + kk * 64 + tn * 16 + l15];
      }
      c = WMMA_BF16(ks == 0 ? a0 : a1, bfg, c);
    }
#pragma unroll
    for (int r = 0; r < 8; r++) {
      int rowi = row0 + r + 8 * half;
      qm[(size_t)((b * 16 + h) * 2048 + rowi) * 64 + tn * 16 + l15] = (__bf16)c[r];
    }
  }
}

// ---------------- flash attention: scores=qm@K^T/8, causal softmax, @V ----------------
__global__ void k_attn(const __bf16* __restrict__ qm, const __bf16* __restrict__ projbf,
                       __bf16* __restrict__ nud) {
  __shared__ __align__(16) __bf16 KV[8][32 * 64];
  __shared__ __align__(16) __bf16 PT[8][16 * 32];
  int lane = threadIdx.x & 31, wid = threadIdx.x >> 5;
  int half = lane >> 4, l15 = lane & 15;
  int blk = blockIdx.x;            // 512
  int qs = blk & 15;
  int h = (blk >> 4) & 15;
  int b = blk >> 8;
  int qbase = (qs * 8 + wid) * 16;

  const __bf16* qrow = qm + (size_t)((b * 16 + h) * 2048 + qbase + l15) * 64;
  v16bf aq0 = make_frag(qrow + half * 8,      qrow + 16 + half * 8);
  v16bf aq1 = make_frag(qrow + 32 + half * 8, qrow + 48 + half * 8);

  __bf16* kv = KV[wid];
  __bf16* pt = PT[wid];
  const __bf16* vbase = projbf + (size_t)(b * 2048) * 1024 + h * 64;

  float mrow[8], lrow[8];
  v8f acc[4] = {v8f{}, v8f{}, v8f{}, v8f{}};
#pragma unroll
  for (int r = 0; r < 8; r++) { mrow[r] = -3.0e38f; lrow[r] = 0.f; }

  int jbmax = (qbase + 15) >> 5;
  for (int jb = 0; jb <= jbmax; jb++) {
    // stage 32 K/V rows (head slice) into per-wave LDS
    {
      const __bf16* src = vbase + (size_t)(jb * 32 + lane) * 1024;
      __bf16* dst = kv + lane * 64;
#if defined(HAVE_ASYNC_LDS)
#pragma unroll
      for (int i = 0; i < 8; i++)
        __builtin_amdgcn_global_load_async_to_lds_b128(
            (__attribute__((address_space(1))) v4i*)(src + i * 8),
            (__attribute__((address_space(3))) v4i*)(dst + i * 8), 0, 0);
      __builtin_amdgcn_s_wait_asynccnt(0);
#else
#pragma unroll
      for (int i = 0; i < 8; i++)
        *(v8bf*)(dst + i * 8) = *(const v8bf*)(src + i * 8);
#endif
    }
    // two 16-col score tiles, K=64
    v8f S[2];
#pragma unroll
    for (int t = 0; t < 2; t++) {
      const __bf16* kr = kv + (t * 16 + l15) * 64;
      v16bf bk0 = make_frag(kr + half * 8,      kr + 16 + half * 8);
      v16bf bk1 = make_frag(kr + 32 + half * 8, kr + 48 + half * 8);
      v8f s = {};
      s = WMMA_BF16(aq0, bk0, s);
      s = WMMA_BF16(aq1, bk1, s);
      S[t] = s;
    }
    bool needMask = (jb * 32 + 31) > qbase;
#pragma unroll
    for (int t = 0; t < 2; t++) {
#pragma unroll
      for (int r = 0; r < 8; r++) {
        float v = S[t][r] * 0.125f;
        if (needMask) {
          int rowi = qbase + r + 8 * half;
          int colg = jb * 32 + t * 16 + l15;
          if (colg > rowi) v = -3.0e38f;
        }
        S[t][r] = v;
      }
    }
    // online softmax over 32 cols; row lives on 16 lanes of one half-wave
#pragma unroll
    for (int r = 0; r < 8; r++) {
      float mx = fmaxf(S[0][r], S[1][r]);
      for (int d = 1; d < 16; d <<= 1) mx = fmaxf(mx, __shfl_xor(mx, d, 32));
      float mnew = fmaxf(mrow[r], mx);
      float alpha = __expf(mrow[r] - mnew);
      float p0 = __expf(S[0][r] - mnew);
      float p1 = __expf(S[1][r] - mnew);
      float ps = p0 + p1;
      for (int d = 1; d < 16; d <<= 1) ps += __shfl_xor(ps, d, 32);
      lrow[r] = lrow[r] * alpha + ps;
      mrow[r] = mnew;
#pragma unroll
      for (int t2 = 0; t2 < 4; t2++) acc[t2][r] *= alpha;
      pt[(r + 8 * half) * 32 + l15]      = (__bf16)p0;
      pt[(r + 8 * half) * 32 + 16 + l15] = (__bf16)p1;
    }
    // re-layout P (C/D layout -> A fragment) via LDS
    const __bf16* pr = pt + l15 * 32;
    v16bf ap = make_frag(pr + half * 8, pr + 16 + half * 8);
    // att @ V, K=32
#pragma unroll
    for (int t = 0; t < 4; t++) {
      v16bf bv;
#pragma unroll
      for (int e = 0; e < 16; e++) {
        int jr = (e < 8) ? (half * 8 + e) : (16 + half * 8 + (e - 8));
        bv[e] = kv[jr * 64 + t * 16 + l15];
      }
      acc[t] = WMMA_BF16(ap, bv, acc[t]);
    }
  }
  // normalize, store nudged in [b,w,c] bf16
#pragma unroll
  for (int r = 0; r < 8; r++) {
    float inv = 1.f / lrow[r];
    int rowi = qbase + r + 8 * half;
#pragma unroll
    for (int t = 0; t < 4; t++)
      nud[(size_t)(b * 2048 + rowi) * 1024 + h * 64 + t * 16 + l15] = (__bf16)(acc[t][r] * inv);
  }
}

extern "C" void kernel_launch(void* const* d_in, const int* in_sizes, int n_in,
                              void* d_out, int out_size, void* d_ws, size_t ws_size,
                              hipStream_t stream) {
  const float* x     = (const float*)d_in[0];  // [2,2048,1024]
  const float* Wproj = (const float*)d_in[1];  // [1024,1024]
  const float* pmet  = (const float*)d_in[2];  // [16,64,64]
  const float* Wmix  = (const float*)d_in[3];  // [1024,1024]
  float* out = (float*)d_out;                  // [2,2048,1024]

  char* ws = (char*)d_ws;
  const size_t MB = 1u << 20;
  __bf16* xbf    = (__bf16*)(ws);             //  8 MB: x bf16
  __bf16* projbf = (__bf16*)(ws +  8 * MB);   //  8 MB: proj bf16
  __bf16* qmbf   = (__bf16*)(ws + 16 * MB);   //  8 MB: qm bf16
  __bf16* nudbf  = (__bf16*)(ws + 24 * MB);   //  8 MB: nudged bf16
  __bf16* BsP    = (__bf16*)(ws + 32 * MB);   //  2 MB: W_proj^T frag-major
  __bf16* BsM    = (__bf16*)(ws + 34 * MB);   //  2 MB: W_mix^T frag-major
  __bf16* met    = (__bf16*)(ws + 36 * MB);   // 128KB: metric bf16

  const int NELEM = 2 * 2048 * 1024;  // 4M

  k_cvt_bf16<<<(NELEM + 255) / 256, 256, 0, stream>>>(x, xbf, NELEM);
  k_shuffle_w<<<4096, 256, 0, stream>>>(Wproj, BsP);
  k_shuffle_w<<<4096, 256, 0, stream>>>(Wmix, BsM);
  k_metric<<<16, 64, 0, stream>>>(pmet, met);

  // proj = x @ W_proj^T  -> bf16
  k_gemm<<<dim3(16, 32), 256, 0, stream>>>(xbf, BsP, nullptr, projbf);
  // qm = proj @ metric (per head)
  k_qm<<<512, 256, 0, stream>>>(projbf, met, qmbf);
  // flash attention -> nudged bf16 [b,w,c]
  k_attn<<<512, 256, 0, stream>>>(qmbf, projbf, nudbf);
  // out = nudged @ W_mix^T -> fp32
  k_gemm<<<dim3(16, 32), 256, 0, stream>>>(nudbf, BsM, out, nullptr);
}
